// SingleQMSANHead_33767032881413
// MI455X (gfx1250) — compile-verified, matrix operand order checked by reading, not run on hardware
//
#include <hip/hip_runtime.h>

typedef __attribute__((ext_vector_type(16))) _Float16 v16h;
typedef __attribute__((ext_vector_type(8)))  _Float16 v8h;
typedef __attribute__((ext_vector_type(8)))  float    v8f;

#define NB 16
#define NS 1024
#define NE 8
#define NZ 256
#define KDIM 512                          // cos(256) || sin(256)
#define TOKENS (NB * NS)                  // 16384 tokens per tensor
#define TAB_ELEMS ((size_t)TOKENS * KDIM) // 8,388,608 halves = 16 MB per table

// ---------------------------------------------------------------------------
// Kernel 1: f16 cos/sin phase tables (scaled 1/16 -> GEMM yields Re/Im of ip
// with the 2^-E normalization folded in).
//   At  [q-tok][k] : k<256 -> cos(phi_q)/16 ; k>=256 -> sin(phi_q)/16
//   Aim [q-tok][k] : k<256 -> sin(phi_q)/16 ; k>=256 -> -cos(phi_q)/16
//   Bre [k-tok][k] : k<256 -> cos(phi_k)/16 ; k>=256 -> sin(phi_k)/16
// Identities:  At·Bre^T = Re(ip),  Aim·Bre^T = -Im(ip)  (sign dies in |ip|^2).
// Only ONE table streams on the column side of the GEMM.
// ---------------------------------------------------------------------------
__global__ void __launch_bounds__(256)
iqp_phase_tables(const float* __restrict__ Q,
                 const float* __restrict__ K,
                 _Float16* __restrict__ At,
                 _Float16* __restrict__ Aim,
                 _Float16* __restrict__ Bre) {
  int g = blockIdx.x;
  bool isQ = (g < TOKENS);
  int idx = isQ ? g : (g - TOKENS);
  const float* xp = (isQ ? Q : K) + (size_t)idx * NE;

  int z = threadIdx.x;                       // basis state 0..255
  float L = 0.0f, sq = 0.0f;
#pragma unroll
  for (int i = 0; i < NE; ++i) {
    float xi = xp[i];
    sq += xi * xi;
    L += ((z >> i) & 1) ? -xi : xi;          // s_i = (-1)^{z_i}
  }
  float phase = -0.5f * L - 0.25f * (L * L - sq);
  float sn, cs;
  __sincosf(phase, &sn, &cs);
  float c16 = cs * 0.0625f;
  float s16 = sn * 0.0625f;

  size_t base = (size_t)idx * KDIM;
  if (isQ) {
    At[base + z]       = (_Float16)c16;
    At[base + NZ + z]  = (_Float16)s16;
    Aim[base + z]      = (_Float16)s16;
    Aim[base + NZ + z] = (_Float16)(-c16);
  } else {
    Bre[base + z]      = (_Float16)c16;
    Bre[base + NZ + z] = (_Float16)s16;
  }
}

// Load a 16x32 f16 WMMA operand fragment from a row-major [token][512] table.
// CDNA5 16-bit A layout: lanes 0-15 hold row M=lane with K {k0..k0+7, k0+16..23},
// lanes 16-31 hold row M=lane-16 with K {k0+8..15, k0+24..31}. Two b128 loads.
__device__ __forceinline__ v16h load_frag(const _Float16* __restrict__ tab,
                                          int tok0, int lane, int k0) {
  int r  = tok0 + (lane & 15);
  int hi = (lane >> 4) & 1;
  const _Float16* p = tab + (size_t)r * KDIM + (k0 + hi * 8);
  v16h f;
  *(v8h*)&f       = *(const v8h*)(p);        // elements 0..7
  *((v8h*)&f + 1) = *(const v8h*)(p + 16);   // elements 8..15
  return f;
}

// ---------------------------------------------------------------------------
// Kernel 2: per (batch, 16-row tile). Each of 4 waves sweeps 4 QUADS of
// 16-col tiles. Per k-step: 2 At + 2 Aim + 8 Bre b128 loads feed 8 WMMAs
// (1.5 loads/WMMA). k-loop is manually double-buffered so WMMAs run behind
// outstanding loads. attn=2(Re^2+Im^2)-1 fused with the 16x8 V tile through
// a per-wave LDS region; partials reduced with LDS float atomics.
// ---------------------------------------------------------------------------
__global__ void __launch_bounds__(128)
qmsan_wmma_attnv(const float* __restrict__ V,
                 const _Float16* __restrict__ At,
                 const _Float16* __restrict__ Aim,
                 const _Float16* __restrict__ Bre,
                 float* __restrict__ out) {
  __shared__ float lds_attn[4 * 16 * 17];    // per-wave 16x16 tile, padded
  __shared__ float outbuf[16 * NE];          // 16 rows x 8 embed

  const int b    = blockIdx.y;
  const int ti   = blockIdx.x;
  const int i0   = ti * 16;
  const int wave = threadIdx.x >> 5;
  const int lane = threadIdx.x & 31;

  outbuf[threadIdx.x] = 0.0f;
  __syncthreads();

  const int    qtok0 = b * NS + i0;
  const float* Vb    = V + (size_t)b * NS * NE;
  float*       my    = lds_attn + wave * (16 * 17);

  const int col   = lane & 15;               // C-layout: N = lane&15
  const int rbase = (lane >> 4) * 8;         // C-layout: M = vgpr + 8*(lane>>4)
  const int i     = lane & 15;               // matvec row owned by this lane
  const int eb    = (lane >> 4) * 4;         // embed half owned by this lane

  float o0 = 0.0f, o1 = 0.0f, o2 = 0.0f, o3 = 0.0f;

  for (int it = 0; it < 4; ++it) {
    const int jt0   = (it * 4 + wave) * 4;   // this wave's tile quad
    const int ktok0 = b * NS + jt0 * 16;

    v8f acc_re[4] = {{}, {}, {}, {}};
    v8f acc_im[4] = {{}, {}, {}, {}};

    // Prologue: fragments for k-step 0.
    v16h aR_c = load_frag(At,  qtok0, lane, 0);
    v16h aI_c = load_frag(Aim, qtok0, lane, 0);
    v16h b_c[4];
#pragma unroll
    for (int t = 0; t < 4; ++t)
      b_c[t] = load_frag(Bre, ktok0 + 16 * t, lane, 0);

#pragma unroll
    for (int ks = 0; ks < 16; ++ks) {
      // Issue next k-step's loads BEFORE consuming current fragments.
      v16h aR_n = aR_c, aI_n = aI_c;
      v16h b_n[4] = {b_c[0], b_c[1], b_c[2], b_c[3]};
      if (ks < 15) {
        const int k0 = (ks + 1) * 32;
        aR_n = load_frag(At,  qtok0, lane, k0);
        aI_n = load_frag(Aim, qtok0, lane, k0);
#pragma unroll
        for (int t = 0; t < 4; ++t)
          b_n[t] = load_frag(Bre, ktok0 + 16 * t, lane, k0);
      }
#pragma unroll
      for (int t = 0; t < 4; ++t) {
        acc_re[t] = __builtin_amdgcn_wmma_f32_16x16x32_f16(
            false, aR_c, false, b_c[t], (short)0, acc_re[t], false, false);
        acc_im[t] = __builtin_amdgcn_wmma_f32_16x16x32_f16(
            false, aI_c, false, b_c[t], (short)0, acc_im[t], false, false);
      }
      aR_c = aR_n; aI_c = aI_n;
#pragma unroll
      for (int t = 0; t < 4; ++t) b_c[t] = b_n[t];
    }

    // Epilogue for each tile of the quad: attn tile -> LDS -> fused x V.
#pragma unroll
    for (int t = 0; t < 4; ++t) {
#pragma unroll
      for (int v = 0; v < 8; ++v) {
        float re = acc_re[t][v], im = acc_im[t][v];
        my[(rbase + v) * 17 + col] = 2.0f * (re * re + im * im) - 1.0f;
      }
      asm volatile("s_wait_dscnt 0x0" ::: "memory");  // CDNA5 split DS counter

      const int j0 = (jt0 + t) * 16;
#pragma unroll
      for (int j = 0; j < 16; ++j) {
        float  aij = my[i * 17 + j];
        float4 vv  = *reinterpret_cast<const float4*>(
            Vb + (size_t)(j0 + j) * NE + eb);
        o0 += aij * vv.x;
        o1 += aij * vv.y;
        o2 += aij * vv.z;
        o3 += aij * vv.w;
      }
    }
  }

  // cross-wave reduction in LDS (ds_add_f32)
  atomicAdd(&outbuf[i * NE + eb + 0], o0);
  atomicAdd(&outbuf[i * NE + eb + 1], o1);
  atomicAdd(&outbuf[i * NE + eb + 2], o2);
  atomicAdd(&outbuf[i * NE + eb + 3], o3);
  __syncthreads();

  const int r = threadIdx.x >> 3;            // 0..15
  const int e = threadIdx.x & 7;             // 0..7
  out[((size_t)b * NS + i0 + r) * NE + e] = outbuf[threadIdx.x];
}

// ---------------------------------------------------------------------------
extern "C" void kernel_launch(void* const* d_in, const int* in_sizes, int n_in,
                              void* d_out, int out_size, void* d_ws, size_t ws_size,
                              hipStream_t stream) {
  const float* Q = (const float*)d_in[0];   // [16,1024,8]
  const float* K = (const float*)d_in[1];
  const float* V = (const float*)d_in[2];
  float* out = (float*)d_out;               // [16,1024,8]

  _Float16* At  = (_Float16*)d_ws;          // 16 MB
  _Float16* Aim = At  + TAB_ELEMS;          // 16 MB
  _Float16* Bre = Aim + TAB_ELEMS;          // 16 MB  (needs ws >= 48 MB)

  iqp_phase_tables<<<2 * TOKENS, 256, 0, stream>>>(Q, K, At, Aim, Bre);

  dim3 grid(NS / 16, NB);                   // (64 row-tiles, 16 batches)
  qmsan_wmma_attnv<<<grid, 128, 0, stream>>>(V, At, Aim, Bre, out);
}
